// PointFeatureConv_49435073577075
// MI455X (gfx1250) — matrix operand order repaired and checked by reading, not compiled, not run on hardware
//
#include <hip/hip_runtime.h>
#include <hip/hip_bf16.h>
#include <math.h>

// ---------------------------------------------------------------------------
// PointFeatureConv for MI455X (gfx1250, wave32, WMMA f32_16x16x32_f16 + TDM)
//   B=2, N=16384, Cin=32, Cout=64, H=128, K=16, Ein=67 (pad->96)
// Kernel 0: pack weights -> f16 fragment-order arrays in workspace
// Kernel 1: brute-force KNN (LDS-tiled, register top-16)
// Kernel 2: fused edge MLP, 1 wave = 1 point, 8 points/wave loop,
//           weights staged to LDS via tensor_load_to_lds (TDM)
// Kernel 3: output MLP, 1 wave = 16 points, 4 tiles/wave loop, TDM staging
// ---------------------------------------------------------------------------

#define BATCH 2
#define NPTS  16384
#define CIN   32
#define COUT  64
#define HID   128
#define KNN   16
#define EIN   67
#define EKP   96        // Ein padded to 3 x 32

#define W1P_ELEMS 12288   // 96x128  (24 frags x 512)
#define W2P_ELEMS 8192    // 128x64  (16 frags x 512)
#define W3P_ELEMS 8192    // 64x128  (16 frags x 512)
#define W4P_ELEMS 8192    // 128x64  (16 frags x 512)

typedef __attribute__((ext_vector_type(16))) _Float16 v16h;
typedef __attribute__((ext_vector_type(8)))  _Float16 v8h;
typedef __attribute__((ext_vector_type(8)))  float    v8f;
typedef __attribute__((ext_vector_type(4)))  unsigned int u32x4;
typedef __attribute__((ext_vector_type(8)))  unsigned int u32x8;

__device__ __forceinline__ float gelu_tanh(float x) {
    const float c = 0.7978845608028654f;           // sqrt(2/pi)
    float t = tanhf(c * (x + 0.044715f * x * x * x));
    return 0.5f * x * (1.0f + t);
}

// Load a 16x32 f16 A-fragment from a row-major LDS tile (row stride ldk elems).
// lanes 0-15 (m=lane) hold K {k0..k0+15}; lanes 16-31
// hold K {k0+16..k0+31} -> two contiguous 16B LDS loads per lane.
__device__ __forceinline__ v16h load_a_frag(const _Float16* base, int ldk,
                                            int k0, int lane) {
    int m  = lane & 15;
    int kh = (lane >> 4) << 4;                      // 0 or 16
    const _Float16* r0 = base + m * ldk + k0 + kh;
    v8h lo = *(const v8h*)(r0);
    v8h hi = *(const v8h*)(r0 + 8);
    v16h a;
#pragma unroll
    for (int e = 0; e < 8; ++e) { a[e] = lo[e]; a[e + 8] = hi[e]; }
    return a;
}

// Issue one TDM copy: nbytes (multiple of 8) from global gsrc -> LDS ldst.
// D# group0/1 per ISA 8.3/8.4: 1-D tile, data_size=8B, tile_dim0=nbytes/8.
// Caller must be a single wave; follow with s_wait_tensorcnt(0) + barrier.
__device__ __forceinline__ void tdm_load_to_lds(const void* gsrc, void* ldst,
                                                unsigned nbytes) {
    unsigned long long ga = (unsigned long long)(uintptr_t)gsrc;
    unsigned lds_off = (unsigned)(uintptr_t)ldst;    // low 32b of flat = LDS offset
    unsigned n8 = nbytes >> 3;                       // elements of 8 bytes
    u32x4 g0;
    g0.x = 1u;                                       // count=1, user mode
    g0.y = lds_off;                                  // lds_addr
    g0.z = (unsigned)ga;                             // global_addr[31:0]
    g0.w = ((unsigned)(ga >> 32) & 0x01FFFFFFu) | (2u << 30);  // addr[56:32]|type=2
    u32x8 g1;
    g1.s0 = 3u << 16;                                // data_size = 8 bytes
    g1.s1 = (n8 & 0xFFFFu) << 16;                    // tensor_dim0[15:0]
    g1.s2 = ((n8 >> 16) & 0xFFFFu) | (1u << 16);     // dim0 hi | tensor_dim1=1
    g1.s3 = (n8 & 0xFFFFu) << 16;                    // dim1 hi=0 | tile_dim0=n8
    g1.s4 = 1u;                                      // tile_dim1=1, tile_dim2=0
    g1.s5 = n8;                                      // tensor_dim0_stride lo
    g1.s6 = 0u;
    g1.s7 = 0u;
    asm volatile("tensor_load_to_lds %0, %1" :: "s"(g0), "s"(g1) : "memory");
}

// ---------------------------------------------------------------------------
// Kernel 0: pack W1..W4 (f32, k-major) into f16 fragment-lane order so every
// WMMA B operand becomes one contiguous 32B LDS read.
//   B layout: frag entry ((frag*32 + lane)*16 + e), k = kc*32 + e + (lane>=16)*16
// ---------------------------------------------------------------------------
__global__ void __launch_bounds__(256)
pack_weights_kernel(const float* __restrict__ W1, const float* __restrict__ W2,
                    const float* __restrict__ W3, const float* __restrict__ W4,
                    _Float16* __restrict__ wp) {
    int i = blockIdx.x * 256 + threadIdx.x;          // 0 .. 36863
    int base, j;
    if (i < W1P_ELEMS) {
        j = i;
        int frag = j >> 9, l = (j >> 4) & 31, e = j & 15;
        int t = frag / 3, kc = frag % 3;
        int k = kc * 32 + e + ((l >> 4) << 4);
        int n = t * 16 + (l & 15);
        wp[j] = (_Float16)((k < EIN) ? W1[k * HID + n] : 0.0f);
    } else if (i < W1P_ELEMS + W2P_ELEMS) {
        base = W1P_ELEMS; j = i - base;
        int frag = j >> 9, l = (j >> 4) & 31, e = j & 15;
        int t = frag >> 2, kc = frag & 3;
        int k = kc * 32 + e + ((l >> 4) << 4);
        int n = t * 16 + (l & 15);
        wp[base + j] = (_Float16)W2[k * COUT + n];
    } else if (i < W1P_ELEMS + W2P_ELEMS + W3P_ELEMS) {
        base = W1P_ELEMS + W2P_ELEMS; j = i - base;
        int frag = j >> 9, l = (j >> 4) & 31, e = j & 15;
        int t = frag >> 1, kc = frag & 1;
        int k = kc * 32 + e + ((l >> 4) << 4);
        int n = t * 16 + (l & 15);
        wp[base + j] = (_Float16)W3[k * HID + n];
    } else {
        base = W1P_ELEMS + W2P_ELEMS + W3P_ELEMS; j = i - base;
        int frag = j >> 9, l = (j >> 4) & 31, e = j & 15;
        int t = frag >> 2, kc = frag & 3;
        int k = kc * 32 + e + ((l >> 4) << 4);
        int n = t * 16 + (l & 15);
        wp[base + j] = (_Float16)W4[k * COUT + n];
    }
}

// ---------------------------------------------------------------------------
// Kernel 1: KNN.  256 threads/block, one query point per thread.
// ---------------------------------------------------------------------------
#define KNN_TILE 2048

__global__ void __launch_bounds__(256)
knn_kernel(const float* __restrict__ positions, int* __restrict__ idx_out) {
    __shared__ float sP[KNN_TILE * 3];

    const int b  = blockIdx.x >> 6;                  // 64 blocks per batch
    const int q  = ((blockIdx.x & 63) << 8) + threadIdx.x;
    const float* pb = positions + (size_t)b * NPTS * 3;

    const float qx = pb[q * 3 + 0];
    const float qy = pb[q * 3 + 1];
    const float qz = pb[q * 3 + 2];

    float bd[KNN];
    int   bi[KNN];
#pragma unroll
    for (int m = 0; m < KNN; ++m) { bd[m] = 3.4e38f; bi[m] = 0; }

    for (int t = 0; t < NPTS / KNN_TILE; ++t) {
        __syncthreads();
        const float* src = pb + t * KNN_TILE * 3;
        for (int i = threadIdx.x; i < KNN_TILE * 3; i += 256) sP[i] = src[i];
        __syncthreads();

        for (int j = 0; j < KNN_TILE; ++j) {
            float dx = qx - sP[3 * j + 0];
            float dy = qy - sP[3 * j + 1];
            float dz = qz - sP[3 * j + 2];
            float d  = dx * dx + dy * dy + dz * dz;
            if (d < bd[KNN - 1]) {
                int jj = t * KNN_TILE + j;
#pragma unroll
                for (int m = KNN - 1; m > 0; --m) {
                    bool mv = bd[m - 1] > d;
                    bd[m] = mv ? bd[m - 1] : bd[m];
                    bi[m] = mv ? bi[m - 1] : bi[m];
                }
#pragma unroll
                for (int m = 0; m < KNN; ++m) {
                    bool ins = (bd[m] > d) && (m == 0 || bd[m - 1] <= d);
                    if (ins) { bd[m] = d; bi[m] = jj; }
                }
            }
        }
    }

    int* dst = idx_out + ((size_t)b * NPTS + q) * KNN;
#pragma unroll
    for (int m = 0; m < KNN; ++m) dst[m] = bi[m];
}

// ---------------------------------------------------------------------------
// Kernel 2: fused edge MLP + mean.  128 threads = 4 waves; each wave loops
// over 8 points.  Packed W1+W2 (40 KB) staged once per block via TDM.
// ---------------------------------------------------------------------------
#define PPW2 8

__global__ void __launch_bounds__(128)
edge_mlp_kernel(const float* __restrict__ positions,
                const float* __restrict__ features,
                const float* __restrict__ b1, const float* __restrict__ b2,
                const _Float16* __restrict__ wpack,   // W1p(12288) ++ W2p(8192)
                const int*   __restrict__ knn_idx,
                float* __restrict__ agg_out) {
    __shared__ _Float16 sW[W1P_ELEMS + W2P_ELEMS];   // 40 KB
    __shared__ float    sb1[HID];
    __shared__ float    sb2[COUT];
    __shared__ _Float16 sBuf[4][16 * HID];           // per-wave edge / h tile

    const int wave = threadIdx.x >> 5;
    const int lane = threadIdx.x & 31;

    if (wave == 0) {
        tdm_load_to_lds(wpack, &sW[0], (W1P_ELEMS + W2P_ELEMS) * 2);
        __builtin_amdgcn_s_wait_tensorcnt(0);
    }
    if (threadIdx.x < HID)  sb1[threadIdx.x] = b1[threadIdx.x];
    if (threadIdx.x < COUT) sb2[threadIdx.x] = b2[threadIdx.x];
    __syncthreads();

    _Float16* bufw = sBuf[wave];
    const _Float16* sW1 = &sW[0];
    const _Float16* sW2 = &sW[W1P_ELEMS];

    const int r     = lane & 15;                     // neighbor row
    const int half  = lane >> 4;                     // column half
    const int nloc  = lane & 15;
    const int mbase = (lane >> 4) << 3;
    const int pbase = (blockIdx.x * 4 + wave) * PPW2;

#pragma unroll 1
    for (int it = 0; it < PPW2; ++it) {
        const int gpt = pbase + it;
        const int b   = gpt >> 14;
        const int p   = gpt & (NPTS - 1);

        // ---- gather: 16x96 f16 edge tile ----
        const int nbr = knn_idx[((size_t)b * NPTS + p) * KNN + r];
        const float* fn = features  + ((size_t)b * NPTS + nbr) * CIN;
        const float* fs = features  + ((size_t)b * NPTS + p)   * CIN;
        const float* pn = positions + ((size_t)b * NPTS + nbr) * 3;
        const float* ps = positions + ((size_t)b * NPTS + p)   * 3;
        {
            _Float16* erow = bufw + r * EKP;
            int cb = half * 48;
#pragma unroll
            for (int j = 0; j < 48; ++j) {
                int c = cb + j;
                float v;
                if (c < CIN)          v = fn[c];
                else if (c < 2 * CIN) v = fs[c - CIN];
                else if (c < EIN)     v = pn[c - 2 * CIN] - ps[c - 2 * CIN];
                else                  v = 0.0f;
                erow[c] = (_Float16)v;
            }
        }
        __syncthreads();

        // ---- layer 1: edge[16x96] @ W1[96x128] ----
        v8f acc[8];
#pragma unroll
        for (int t = 0; t < 8; ++t) acc[t] = (v8f)0.0f;
#pragma unroll
        for (int kc = 0; kc < 3; ++kc) {
            v16h a = load_a_frag(bufw, EKP, kc * 32, lane);
            v16h bf[8];
#pragma unroll
            for (int t = 0; t < 8; ++t)
                bf[t] = *(const v16h*)&sW1[((t * 3 + kc) * 32 + lane) * 16];
#pragma unroll
            for (int t = 0; t < 8; ++t)
                acc[t] = __builtin_amdgcn_wmma_f32_16x16x32_f16(
                    false, a, false, bf[t], (short)0, acc[t], false, false);
        }
        __syncthreads();                             // edge reads complete

        // ---- bias + gelu -> h tile (f16, ldk=128) in the same buffer ----
#pragma unroll
        for (int t = 0; t < 8; ++t) {
            int c = t * 16 + nloc;
            float bias = sb1[c];
#pragma unroll
            for (int rr = 0; rr < 8; ++rr) {
                float g = gelu_tanh(acc[t][rr] + bias);
                bufw[(mbase + rr) * HID + c] = (_Float16)g;
            }
        }
        __syncthreads();

        // ---- layer 2: h[16x128] @ W2[128x64] ----
        v8f acc2[4];
#pragma unroll
        for (int t = 0; t < 4; ++t) acc2[t] = (v8f)0.0f;
#pragma unroll
        for (int kc = 0; kc < 4; ++kc) {
            v16h a = load_a_frag(bufw, HID, kc * 32, lane);
            v16h bf[4];
#pragma unroll
            for (int t = 0; t < 4; ++t)
                bf[t] = *(const v16h*)&sW2[((t * 4 + kc) * 32 + lane) * 16];
#pragma unroll
            for (int t = 0; t < 4; ++t)
                acc2[t] = __builtin_amdgcn_wmma_f32_16x16x32_f16(
                    false, a, false, bf[t], (short)0, acc2[t], false, false);
        }

        // ---- mean over 16 neighbor rows, + b2, store agg ----
#pragma unroll
        for (int t = 0; t < 4; ++t) {
            int c = t * 16 + nloc;
            float s = 0.0f;
#pragma unroll
            for (int rr = 0; rr < 8; ++rr) s += acc2[t][rr];
            s += __shfl_xor(s, 16, 32);
            float aggv = s * (1.0f / 16.0f) + sb2[c];
            if (lane < 16)
                agg_out[((size_t)b * NPTS + p) * COUT + c] = aggv;
        }
        __syncthreads();                             // h buffer free for next point
    }
}

// ---------------------------------------------------------------------------
// Kernel 3: output MLP.  128 threads = 4 waves; each wave loops over 4 tiles
// of 16 points.  Packed W3+W4 (32 KB) staged once per block via TDM.
// ---------------------------------------------------------------------------
#define TPW3 4

__global__ void __launch_bounds__(128)
out_mlp_kernel(const float* __restrict__ agg_in,
               const float* __restrict__ b3, const float* __restrict__ b4,
               const _Float16* __restrict__ wpack,   // W3p(8192) ++ W4p(8192)
               float* __restrict__ out) {
    __shared__ _Float16 sW[W3P_ELEMS + W4P_ELEMS];   // 32 KB
    __shared__ float    sb3[HID];
    __shared__ float    sb4[COUT];
    __shared__ _Float16 sBuf[4][16 * HID];

    const int wave = threadIdx.x >> 5;
    const int lane = threadIdx.x & 31;

    if (wave == 0) {
        tdm_load_to_lds(wpack, &sW[0], (W3P_ELEMS + W4P_ELEMS) * 2);
        __builtin_amdgcn_s_wait_tensorcnt(0);
    }
    if (threadIdx.x < HID)  sb3[threadIdx.x] = b3[threadIdx.x];
    if (threadIdx.x < COUT) sb4[threadIdx.x] = b4[threadIdx.x];
    __syncthreads();

    _Float16* bufw = sBuf[wave];
    const _Float16* sW3 = &sW[0];
    const _Float16* sW4 = &sW[W3P_ELEMS];

    const int nloc  = lane & 15;
    const int mbase = (lane >> 4) << 3;
    const int tbase = (blockIdx.x * 4 + wave) * TPW3;

#pragma unroll 1
    for (int it = 0; it < TPW3; ++it) {
        const size_t p0 = (size_t)(tbase + it) * 16;

        // stage A = agg[16 x 64] as f16 (ldk = 64)
        {
            int rr = lane & 15, cb = (lane >> 4) * 32;
            const float* src = agg_in + (p0 + rr) * COUT;
#pragma unroll
            for (int j = 0; j < 32; ++j)
                bufw[rr * COUT + cb + j] = (_Float16)src[cb + j];
        }
        __syncthreads();

        // layer 3: agg[16x64] @ W3[64x128]
        v8f acc[8];
#pragma unroll
        for (int t = 0; t < 8; ++t) acc[t] = (v8f)0.0f;
#pragma unroll
        for (int kc = 0; kc < 2; ++kc) {
            v16h a = load_a_frag(bufw, COUT, kc * 32, lane);
            v16h bf[8];
#pragma unroll
            for (int t = 0; t < 8; ++t)
                bf[t] = *(const v16h*)&sW3[((t * 2 + kc) * 32 + lane) * 16];
#pragma unroll
            for (int t = 0; t < 8; ++t)
                acc[t] = __builtin_amdgcn_wmma_f32_16x16x32_f16(
                    false, a, false, bf[t], (short)0, acc[t], false, false);
        }
        __syncthreads();

        // bias + gelu -> h tile (ldk = 128)
#pragma unroll
        for (int t = 0; t < 8; ++t) {
            int c = t * 16 + nloc;
            float bias = sb3[c];
#pragma unroll
            for (int rr = 0; rr < 8; ++rr) {
                float g = gelu_tanh(acc[t][rr] + bias);
                bufw[(mbase + rr) * HID + c] = (_Float16)g;
            }
        }
        __syncthreads();

        // layer 4: h[16x128] @ W4[128x64] + b4 -> f32 out
        v8f acc2[4];
#pragma unroll
        for (int t = 0; t < 4; ++t) acc2[t] = (v8f)0.0f;
#pragma unroll
        for (int kc = 0; kc < 4; ++kc) {
            v16h a = load_a_frag(bufw, HID, kc * 32, lane);
            v16h bf[4];
#pragma unroll
            for (int t = 0; t < 4; ++t)
                bf[t] = *(const v16h*)&sW4[((t * 4 + kc) * 32 + lane) * 16];
#pragma unroll
            for (int t = 0; t < 4; ++t)
                acc2[t] = __builtin_amdgcn_wmma_f32_16x16x32_f16(
                    false, a, false, bf[t], (short)0, acc2[t], false, false);
        }
#pragma unroll
        for (int t = 0; t < 4; ++t) {
            int c = t * 16 + nloc;
            float bias = sb4[c];
#pragma unroll
            for (int rr = 0; rr < 8; ++rr)
                out[(p0 + mbase + rr) * COUT + c] = acc2[t][rr] + bias;
        }
        __syncthreads();
    }
}

// ---------------------------------------------------------------------------
extern "C" void kernel_launch(void* const* d_in, const int* in_sizes, int n_in,
                              void* d_out, int out_size, void* d_ws, size_t ws_size,
                              hipStream_t stream) {
    const float* positions = (const float*)d_in[0];
    const float* features  = (const float*)d_in[1];
    const float* W1 = (const float*)d_in[2];
    const float* b1 = (const float*)d_in[3];
    const float* W2 = (const float*)d_in[4];
    const float* b2 = (const float*)d_in[5];
    const float* W3 = (const float*)d_in[6];
    const float* b3 = (const float*)d_in[7];
    const float* W4 = (const float*)d_in[8];
    const float* b4 = (const float*)d_in[9];
    float* out = (float*)d_out;

    // workspace: [idx: B*N*K int][agg: B*N*COUT f32][packed weights: f16]
    char* wsb = (char*)d_ws;
    const size_t idx_bytes = (size_t)BATCH * NPTS * KNN * sizeof(int);     // 2 MB
    const size_t agg_bytes = (size_t)BATCH * NPTS * COUT * sizeof(float);  // 8 MB
    int*      d_idx = (int*)wsb;
    float*    d_agg = (float*)(wsb + idx_bytes);
    _Float16* d_wp  = (_Float16*)(wsb + idx_bytes + agg_bytes);
    _Float16* d_w12 = d_wp;                                  // W1p ++ W2p
    _Float16* d_w34 = d_wp + W1P_ELEMS + W2P_ELEMS;          // W3p ++ W4p

    pack_weights_kernel<<<(W1P_ELEMS + W2P_ELEMS + W3P_ELEMS + W4P_ELEMS) / 256,
                          256, 0, stream>>>(W1, W2, W3, W4, d_wp);

    knn_kernel<<<BATCH * (NPTS / 256), 256, 0, stream>>>(positions, d_idx);

    edge_mlp_kernel<<<(BATCH * NPTS) / (4 * PPW2), 128, 0, stream>>>(
        positions, features, b1, b2, d_w12, d_idx, d_agg);

    out_mlp_kernel<<<(BATCH * NPTS) / (4 * 16 * TPW3), 128, 0, stream>>>(
        d_agg, b3, b4, d_w34, out);
}
